// MLPPredictor_78426102825753
// MI455X (gfx1250) — compile-verified, matrix operand order checked by reading, not run on hardware
//
#include <hip/hip_runtime.h>
#include <hip/hip_bf16.h>

typedef __attribute__((ext_vector_type(16))) __bf16 v16bf;
typedef __attribute__((ext_vector_type(8)))  float  v8f;

#define FDIM 128        // per-table feature dim
#define TWOF 256        // concat dim (K of GEMM)
#define HDIM 128        // hidden dim (N of GEMM)
#define KT   8          // 256 / 32  k-tiles
#define NT   8          // 128 / 16  n-tiles

// ---------------------------------------------------------------------------
// Pre-pack W1 (f32, row-major [256,128]) into bf16 WMMA-B per-lane layout.
// packed[kt*4096 + nt*512 + lane*16 + j]  where for lane<16: col=lane,
// K(j) = kt*32 + (j<8 ? j : j+8); for lane>=16: col=lane-16, K(j) += 8.
// Each lane reads its 16 bf16 (32 B) contiguously from LDS.
// ---------------------------------------------------------------------------
__global__ void pack_w1_kernel(const float* __restrict__ W1,
                               __bf16* __restrict__ out) {
    int t  = blockIdx.x * blockDim.x + threadIdx.x;   // 0 .. 32767
    int j  = t & 15;
    int l  = (t >> 4) & 31;
    int nt = (t >> 9) & 7;
    int kt = (t >> 12) & 7;
    int k  = kt * 32 + ((j < 8) ? j : (j + 8)) + ((l >= 16) ? 8 : 0);
    int n  = nt * 16 + (l & 15);
    out[t] = (__bf16)W1[k * HDIM + n];
}

// Gather + cvt one A operand: 16 bf16 of row `row`, k-range [k0, k0+32)
// per the 16-bit A-matrix VGPR layout (lane half selects K sub-block).
static __device__ __forceinline__ v16bf gather_a(const float* __restrict__ p) {
    float4 f0 = ((const float4*)(p +  0))[0];
    float4 f1 = ((const float4*)(p +  4))[0];
    float4 f2 = ((const float4*)(p + 16))[0];
    float4 f3 = ((const float4*)(p + 20))[0];
    v16bf a;
    a[0]  = (__bf16)f0.x; a[1]  = (__bf16)f0.y;
    a[2]  = (__bf16)f0.z; a[3]  = (__bf16)f0.w;
    a[4]  = (__bf16)f1.x; a[5]  = (__bf16)f1.y;
    a[6]  = (__bf16)f1.z; a[7]  = (__bf16)f1.w;
    a[8]  = (__bf16)f2.x; a[9]  = (__bf16)f2.y;
    a[10] = (__bf16)f2.z; a[11] = (__bf16)f2.w;
    a[12] = (__bf16)f3.x; a[13] = (__bf16)f3.y;
    a[14] = (__bf16)f3.z; a[15] = (__bf16)f3.w;
    return a;
}

// ---------------------------------------------------------------------------
// Main kernel: 256 threads = 8 waves. Each wave computes 32 edges
// (two 16-row A tiles, register-resident). B streamed once per (kt,nt)
// from LDS and reused by both tiles -> 2 WMMAs per 2 ds_load_b128.
// ---------------------------------------------------------------------------
__global__ void __launch_bounds__(256)
edge_mlp_kernel(const float* __restrict__ hsrc,
                const float* __restrict__ hdst,
                const int*   __restrict__ src,
                const int*   __restrict__ dst,
                const __bf16* __restrict__ w1p,
                const float* __restrict__ b1,
                const float* __restrict__ W2,
                const float* __restrict__ b2,
                float* __restrict__ out) {
    // 64 KB of packed W1 in LDS, shared by all 8 waves of the block.
    __shared__ __align__(16) __bf16 ldsW[TWOF * HDIM];

    const int tid = threadIdx.x;

    // Cooperative stage: 64 KB / (256 threads * 16 B) = 16 uint4 per thread.
    {
        const uint4* g = (const uint4*)w1p;
        uint4* s = (uint4*)ldsW;
#pragma unroll
        for (int i = 0; i < 16; ++i)
            s[tid + 256 * i] = g[tid + 256 * i];
    }
    __syncthreads();

    const int lane   = tid & 31;
    const int wave   = tid >> 5;
    const int pair   = blockIdx.x * 8 + wave;   // 32-edge tile-pair index
    const int row    = lane & 15;               // A-matrix row owned by lane
    const int half   = (lane >> 4) << 3;        // 0 or 8 (K sub-block select)
    const int n_lane = lane & 15;               // B/C column owned by lane
    const int e0     = pair * 32 + row;         // tile 0 edge
    const int e1     = e0 + 16;                 // tile 1 edge

    // Gather ALL A operands once (register-resident): 2 tiles x 8 k-tiles.
    const float* ps0 = hsrc + (long)src[e0] * FDIM;
    const float* pd0 = hdst + (long)dst[e0] * FDIM;
    const float* ps1 = hsrc + (long)src[e1] * FDIM;
    const float* pd1 = hdst + (long)dst[e1] * FDIM;

    v16bf A0[KT], A1[KT];
#pragma unroll
    for (int kt = 0; kt < KT; ++kt) {
        const float* p0 = (kt < 4) ? (ps0 + kt * 32 + half)
                                   : (pd0 + (kt - 4) * 32 + half);
        const float* p1 = (kt < 4) ? (ps1 + kt * 32 + half)
                                   : (pd1 + (kt - 4) * 32 + half);
        A0[kt] = gather_a(p0);
        A1[kt] = gather_a(p1);
    }

    const float bias2 = b2[0];
    float part0[8], part1[8];
#pragma unroll
    for (int r = 0; r < 8; ++r) { part0[r] = 0.0f; part1[r] = 0.0f; }

    // nt outer: only two live accumulators; B tile feeds both A tiles.
#pragma unroll
    for (int nt = 0; nt < NT; ++nt) {
        v8f acc0 = (v8f)(0.0f);
        v8f acc1 = (v8f)(0.0f);
#pragma unroll
        for (int kt = 0; kt < KT; ++kt) {
            union { uint4 u[2]; v16bf v; } bb;
            const uint4* lp =
                (const uint4*)(ldsW + (kt * 8 + nt) * 512 + lane * 16);
            bb.u[0] = lp[0];
            bb.u[1] = lp[1];
            acc0 = __builtin_amdgcn_wmma_f32_16x16x32_bf16(
                false, A0[kt], false, bb.v, (short)0, acc0, false, false);
            acc1 = __builtin_amdgcn_wmma_f32_16x16x32_bf16(
                false, A1[kt], false, bb.v, (short)0, acc1, false, false);
        }
        // Fold epilogue for this nt immediately (kills accumulators).
        const float b1v = b1[nt * 16 + n_lane];
        const float w2v = W2[nt * 16 + n_lane];
#pragma unroll
        for (int r = 0; r < 8; ++r) {
            float h0 = fmaxf(acc0[r] + b1v, 0.0f);
            float h1 = fmaxf(acc1[r] + b1v, 0.0f);
            part0[r] = fmaf(h0, w2v, part0[r]);
            part1[r] = fmaf(h1, w2v, part1[r]);
        }
    }

    // C layout: lanes 0-15 hold rows 0-7, lanes 16-31 hold rows 8-15;
    // xor masks < 16 keep the butterfly within each half.
#pragma unroll
    for (int m = 1; m < 16; m <<= 1) {
#pragma unroll
        for (int r = 0; r < 8; ++r) {
            part0[r] += __shfl_xor(part0[r], m, 32);
            part1[r] += __shfl_xor(part1[r], m, 32);
        }
    }
    if (n_lane == 0) {
#pragma unroll
        for (int r = 0; r < 8; ++r) {
            out[pair * 32      + half + r] = part0[r] + bias2;
            out[pair * 32 + 16 + half + r] = part1[r] + bias2;
        }
    }
}

// ---------------------------------------------------------------------------
extern "C" void kernel_launch(void* const* d_in, const int* in_sizes, int n_in,
                              void* d_out, int out_size, void* d_ws,
                              size_t ws_size, hipStream_t stream) {
    const float* hsrc = (const float*)d_in[0];
    const float* hdst = (const float*)d_in[1];
    const int*   src  = (const int*)d_in[2];
    const int*   dst  = (const int*)d_in[3];
    const float* W1   = (const float*)d_in[4];
    const float* b1   = (const float*)d_in[5];
    const float* W2   = (const float*)d_in[6];
    const float* b2   = (const float*)d_in[7];
    float* out = (float*)d_out;

    __bf16* w1p = (__bf16*)d_ws;   // 64 KB packed W1

    const int E = in_sizes[2];     // 640000

    // Repack W1 each launch (deterministic, cheap: 32768 threads).
    pack_w1_kernel<<<(TWOF * HDIM) / 256, 256, 0, stream>>>(W1, w1p);

    const int pairs  = E / 32;     // 20000 (E is a multiple of 32)
    const int blocks = pairs / 8;  // 2500, 8 waves per block
    edge_mlp_kernel<<<blocks, 256, 0, stream>>>(hsrc, hdst, src, dst, w1p,
                                                b1, W2, b2, out);
}
